// PAM_Module_50646254355175
// MI455X (gfx1250) — matrix operand order repaired and checked by reading, not compiled, not run on hardware
//
#include <hip/hip_runtime.h>
#include <hip/hip_bf16.h>
#include <stdint.h>

typedef __attribute__((ext_vector_type(16))) _Float16 v16h;
typedef __attribute__((ext_vector_type(8)))  float    v8f;

#define BB 4
#define CC 64
#define HH 96
#define WW 96
#define NN (HH*WW)      // 9216
#define HP 48
#define WP 48
#define MM (HP*WP)      // 2304
#define KD 64
#define LOG2E 1.4426950408889634f

union HF {
    v16h v;
    uint4 q[2];
    _Float16 h[16];
};

// 16-lane max butterfly entirely in VALU via DPP16 (co-executes with XDL WMMA).
__device__ __forceinline__ float rowmax16(float v) {
    float t;
    t = __int_as_float(__builtin_amdgcn_update_dpp(0, __float_as_int(v), 0xB1, 0xF, 0xF, true));  // quad_perm xor1
    v = fmaxf(v, t);
    t = __int_as_float(__builtin_amdgcn_update_dpp(0, __float_as_int(v), 0x4E, 0xF, 0xF, true));  // quad_perm xor2
    v = fmaxf(v, t);
    t = __int_as_float(__builtin_amdgcn_update_dpp(0, __float_as_int(v), 0x141, 0xF, 0xF, true)); // row_half_mirror
    v = fmaxf(v, t);
    t = __int_as_float(__builtin_amdgcn_update_dpp(0, __float_as_int(v), 0x140, 0xF, 0xF, true)); // row_mirror
    v = fmaxf(v, t);
    return v;
}

#define WMMA(A, B, C) __builtin_amdgcn_wmma_f32_16x16x32_f16(false, (A), false, (B), (short)0, (C), false, false)

// ---------------------------------------------------------------------------
// Prep 1: maxpool(x), maxpool(c2), edge gate ea (folded into K!), keys
// K*ea (f16, [m][k]), values V (f16, [c][m]).  64-thread block / pooled pixel.
// ---------------------------------------------------------------------------
__global__ void prep_pool_ea_k_v(
    const float* __restrict__ c2, const float* __restrict__ x,
    const float* __restrict__ w_ea1, const float* __restrict__ bn_w,
    const float* __restrict__ bn_b, const float* __restrict__ bn_m,
    const float* __restrict__ bn_v, const float* __restrict__ w_ea2,
    const float* __restrict__ b_ea2, const float* __restrict__ w_q,
    const float* __restrict__ b_q,
    _Float16* __restrict__ k_t, _Float16* __restrict__ v_k)
{
    __shared__ float cat[2*CC];
    __shared__ float pbuf[CC];
    __shared__ float eaSh;
    const int b = blockIdx.x / MM;
    const int m = blockIdx.x % MM;
    const int c = threadIdx.x;
    const int mh = m / WP, mw = m % WP;
    const size_t base = ((size_t)(b*CC + c)*HH + 2*mh)*WW + 2*mw;
    float xp = fmaxf(fmaxf(x[base],  x[base+1]),  fmaxf(x[base+WW],  x[base+WW+1]));
    float cp = fmaxf(fmaxf(c2[base], c2[base+1]), fmaxf(c2[base+WW], c2[base+WW+1]));
    cat[c]      = cp;
    cat[CC + c] = xp;
    v_k[(size_t)(b*CC + c)*MM + m] = (_Float16)xp;
    __syncthreads();

    // h = relu(bn(w_ea1 @ [c2p; xp])) ;  partial for ea
    float acc = 0.f;
    const float* wr = w_ea1 + c*2*CC;
    #pragma unroll 8
    for (int j = 0; j < 2*CC; ++j) acc += wr[j]*cat[j];
    float scale = bn_w[c] * rsqrtf(bn_v[c] + 1e-5f);
    float hv = fmaxf((acc - bn_m[c])*scale + bn_b[c], 0.f);
    pbuf[c] = w_ea2[c]*hv;

    // keys: k = w_q @ xp + b_q (kept in reg; scaled by ea below)
    float ka = b_q[c];
    const float* wq = w_q + c*CC;
    #pragma unroll 8
    for (int j = 0; j < CC; ++j) ka += wq[j]*cat[CC+j];
    __syncthreads();
    if (c == 0) {
        float s = 0.f;
        for (int j = 0; j < CC; ++j) s += pbuf[j];
        s += b_ea2[0];
        eaSh = 1.f/(1.f + __expf(-s));
    }
    __syncthreads();
    k_t[(size_t)(b*MM + m)*KD + c] = (_Float16)(ka * eaSh);
}

// ---------------------------------------------------------------------------
// Prep 2: queries at full res, f16 [n][k]; log2(e) folded in so the softmax
// uses bare v_exp_f32 (exp2).
// ---------------------------------------------------------------------------
__global__ void prep_query(const float* __restrict__ x,
                           const float* __restrict__ w_q,
                           const float* __restrict__ b_q,
                           _Float16* __restrict__ q_t)
{
    __shared__ float xs[4][CC];
    const int p = threadIdx.x / CC;
    const int c = threadIdx.x % CC;
    const int b = blockIdx.x / (NN/4);
    const int n = (blockIdx.x % (NN/4))*4 + p;
    xs[p][c] = x[(size_t)(b*CC + c)*NN + n];
    __syncthreads();
    float acc = b_q[c];
    const float* wq = w_q + c*CC;
    #pragma unroll 8
    for (int j = 0; j < CC; ++j) acc += wq[j]*xs[p][j];
    q_t[(size_t)(b*NN + n)*KD + c] = (_Float16)(acc * LOG2E);
}

// ---------------------------------------------------------------------------
// Fused flash attention.  4 waves/block, 16 query rows per wave, M streamed
// in 64-column chunks: 18 v_wmma per chunk (8 S + 8 PV + 2 rowsum-by-ones).
// Row-max via DPP16 butterflies, ea and log2e pre-folded, row-sum obtained
// as an extra WMMA accumulator (same C/D layout as O -> shuffle-free epilogue).
// ---------------------------------------------------------------------------
__launch_bounds__(128)
__global__ void pam_attention(const _Float16* __restrict__ q_t,
                              const _Float16* __restrict__ k_t,
                              const _Float16* __restrict__ v_k,
                              const float* __restrict__ x,
                              const float* __restrict__ gamma,
                              float* __restrict__ out)
{
    __shared__ _Float16 Pt[4][16][64];   // per-wave P staging (S->A transpose)

    const int wv   = threadIdx.x >> 5;
    const int lane = threadIdx.x & 31;
    const int col  = lane & 15;
    const int hi   = lane >> 4;
    const int b    = blockIdx.x / (NN/64);
    const int n0   = (blockIdx.x % (NN/64))*64 + wv*16;

    // ---- Q A-fragments: 16 rows x 64 K ----
    const _Float16* qrow = q_t + (size_t)(b*NN + n0 + col)*KD;
    const int h0 = hi ? 8 : 0;
    HF aq0, aq1;
    aq0.q[0] = *(const uint4*)(qrow + h0);
    aq0.q[1] = *(const uint4*)(qrow + h0 + 16);
    aq1.q[0] = *(const uint4*)(qrow + h0 + 32);
    aq1.q[1] = *(const uint4*)(qrow + h0 + 48);

    HF ones;
    #pragma unroll
    for (int i = 0; i < 16; ++i) ones.h[i] = (_Float16)1.0f;

    v8f o[4] = {{}, {}, {}, {}};
    v8f osum = {};
    float rmax[8];
    #pragma unroll
    for (int g = 0; g < 8; ++g) rmax[g] = -1e30f;

    const _Float16* kb = k_t + (size_t)b*MM*KD;   // uniform base (SGPR)
    const _Float16* vb = v_k + (size_t)b*CC*MM;   // uniform base (SGPR)
    const uint32_t koff = hi ? 8u : 0u;
    const uint32_t moff = hi ? 8u : 0u;

    for (int mc = 0; mc < MM; mc += 64) {
        // ---- S = Qt K for 4 column tiles (32-bit voffsets from SGPR base) ----
        v8f s[4];
        #pragma unroll
        for (int t = 0; t < 4; ++t) {
            const uint32_t ko = (uint32_t)(mc + t*16 + col)*KD + koff;
            HF bkA, bkB;
            bkA.q[0] = *(const uint4*)(kb + ko);
            bkA.q[1] = *(const uint4*)(kb + ko + 16);
            bkB.q[0] = *(const uint4*)(kb + ko + 32);
            bkB.q[1] = *(const uint4*)(kb + ko + 48);
            v8f acc = {};
            acc = WMMA(aq0.v, bkA.v, acc);
            acc = WMMA(aq1.v, bkB.v, acc);
            s[t] = acc;
        }
        if (mc + 64 < MM) {
            __builtin_prefetch(kb + (uint32_t)(mc + 64 + col)*KD, 0, 1);
            __builtin_prefetch(vb + (uint32_t)col*MM + mc + 64, 0, 1);
        }

        // ---- online softmax over 64 columns (DPP16 max tree, exp2) ----
        #pragma unroll
        for (int g = 0; g < 8; ++g) {
            const float v0 = s[0][g], v1 = s[1][g], v2 = s[2][g], v3 = s[3][g];
            float mx = fmaxf(fmaxf(v0, v1), fmaxf(v2, v3));
            mx = rowmax16(mx);
            const float nm = fmaxf(rmax[g], mx);
            const float alpha = exp2f(rmax[g] - nm);
            rmax[g] = nm;
            o[0][g] = o[0][g]*alpha;  o[1][g] = o[1][g]*alpha;
            o[2][g] = o[2][g]*alpha;  o[3][g] = o[3][g]*alpha;
            osum[g] = osum[g]*alpha;
            const int r = hi ? (g + 8) : g;
            Pt[wv][r][col]      = (_Float16)exp2f(v0 - nm);
            Pt[wv][r][col + 16] = (_Float16)exp2f(v1 - nm);
            Pt[wv][r][col + 32] = (_Float16)exp2f(v2 - nm);
            Pt[wv][r][col + 48] = (_Float16)exp2f(v3 - nm);
        }
        asm volatile("s_wait_dscnt 0" ::: "memory");  // same-wave DS in-order

        HF pf0, pf1;   // P as two 16x32 A-fragments
        pf0.q[0] = *(const uint4*)&Pt[wv][col][h0];
        pf0.q[1] = *(const uint4*)&Pt[wv][col][h0 + 16];
        pf1.q[0] = *(const uint4*)&Pt[wv][col][h0 + 32];
        pf1.q[1] = *(const uint4*)&Pt[wv][col][h0 + 48];

        // ---- O += P V^T ; row-sum rides along as a 5th accumulator ----
        #pragma unroll
        for (int ct = 0; ct < 4; ++ct) {
            const uint32_t vo = (uint32_t)(ct*16 + col)*MM + mc + moff;
            HF bv0, bv1;
            bv0.q[0] = *(const uint4*)(vb + vo);
            bv0.q[1] = *(const uint4*)(vb + vo + 16);
            bv1.q[0] = *(const uint4*)(vb + vo + 32);
            bv1.q[1] = *(const uint4*)(vb + vo + 48);
            o[ct] = WMMA(pf0.v, bv0.v, o[ct]);
            o[ct] = WMMA(pf1.v, bv1.v, o[ct]);
        }
        osum = WMMA(pf0.v, ones.v, osum);
        osum = WMMA(pf1.v, ones.v, osum);
    }

    // ---- epilogue: gamma * O/rowsum + x  (osum shares C/D layout with O) ----
    const float gm = gamma[0];
    #pragma unroll
    for (int g = 0; g < 8; ++g) {
        const int n = n0 + (hi ? g + 8 : g);
        const float inv = gm / osum[g];
        const size_t i0 = ((size_t)(b*CC) + col)*NN + n;
        out[i0]           = o[0][g]*inv + x[i0];
        out[i0 + 16ul*NN] = o[1][g]*inv + x[i0 + 16ul*NN];
        out[i0 + 32ul*NN] = o[2][g]*inv + x[i0 + 32ul*NN];
        out[i0 + 48ul*NN] = o[3][g]*inv + x[i0 + 48ul*NN];
    }
}

// ---------------------------------------------------------------------------
extern "C" void kernel_launch(void* const* d_in, const int* in_sizes, int n_in,
                              void* d_out, int out_size, void* d_ws, size_t ws_size,
                              hipStream_t stream) {
    const float* c2    = (const float*)d_in[0];
    const float* x     = (const float*)d_in[1];
    const float* w_ea1 = (const float*)d_in[2];
    const float* bn_w  = (const float*)d_in[3];
    const float* bn_b  = (const float*)d_in[4];
    const float* bn_m  = (const float*)d_in[5];
    const float* bn_v  = (const float*)d_in[6];
    const float* w_ea2 = (const float*)d_in[7];
    const float* b_ea2 = (const float*)d_in[8];
    const float* w_q   = (const float*)d_in[9];
    const float* b_q   = (const float*)d_in[10];
    const float* gamma = (const float*)d_in[11];
    float* out = (float*)d_out;

    char* ws = (char*)d_ws;
    _Float16* k_t = (_Float16*)ws;
    size_t off = (((size_t)BB*MM*KD*sizeof(_Float16)) + 255) & ~(size_t)255;
    _Float16* v_k = (_Float16*)(ws + off);
    off += (((size_t)BB*CC*MM*sizeof(_Float16)) + 255) & ~(size_t)255;
    _Float16* q_t = (_Float16*)(ws + off);

    prep_pool_ea_k_v<<<BB*MM, CC, 0, stream>>>(c2, x, w_ea1, bn_w, bn_b, bn_m,
                                               bn_v, w_ea2, b_ea2, w_q, b_q,
                                               k_t, v_k);
    prep_query<<<BB*(NN/4), 4*CC, 0, stream>>>(x, w_q, b_q, q_t);
    pam_attention<<<BB*(NN/64), 128, 0, stream>>>(q_t, k_t, v_k, x, gamma, out);
}
